// TriAttnFullWrapper_83708912599799
// MI455X (gfx1250) — matrix operand order repaired and checked by reading, not compiled, not run on hardware
//
#include <hip/hip_runtime.h>
#include <hip/hip_bf16.h>
#include <cstdint>

// ---------------------------------------------------------------------------
// Triangle attention, MI455X (gfx1250, wave32, WMMA bf16 16x16x32 + TDM)
// ---------------------------------------------------------------------------
#define NPOS   384
#define CZ     128
#define NHEADS 4
#define CHID   32
#define PTOT   (NPOS * NPOS)          // 147456 positions
#define EPSLN  1e-5f
#define QSCALE 0.17677669529663688f   // 32^-0.5

typedef __bf16 bf16_t;
typedef __attribute__((ext_vector_type(16))) __bf16 v16bf;
typedef __attribute__((ext_vector_type(8)))  float  v8f;
typedef __attribute__((ext_vector_type(4)))  __bf16 v4bf;

#if defined(__AMDGCN__) && __has_builtin(__builtin_amdgcn_tensor_load_to_lds)
#define USE_TDM 1
typedef unsigned int u32x4 __attribute__((ext_vector_type(4)));
typedef int          i32x4 __attribute__((ext_vector_type(4)));
typedef int          i32x8 __attribute__((ext_vector_type(8)));
#endif

union FragBF {            // one WMMA 16-element bf16 fragment = 2 x 16B
    v16bf v;
    uint4 q[2];
};
union Pack8 {             // 8 bf16 = 16B
    uint4  q;
    bf16_t e[8];
};

__device__ __forceinline__ v8f wmma_bf16(v16bf a, v16bf b, v8f c) {
    return __builtin_amdgcn_wmma_f32_16x16x32_bf16(
        false, a, false, b, (short)0, c, false, false);
}

// ---------------------------------------------------------------------------
// Kernel 0: weight prep — f32 (128x128 row-major) -> bf16 transposed
// ---------------------------------------------------------------------------
__global__ void wt_kernel(const float* __restrict__ W, bf16_t* __restrict__ WT) {
    const int e = blockIdx.x * 256 + threadIdx.x;   // 16384 threads
    const int k = e >> 7, n = e & 127;
    WT[n * 128 + k] = (bf16_t)W[e];
}

// ---------------------------------------------------------------------------
// Kernel 1: LayerNorm over last dim (128), fp32 -> bf16.  One wave per row.
// ---------------------------------------------------------------------------
__global__ void ln_kernel(const float* __restrict__ x,
                          const float* __restrict__ w,
                          const float* __restrict__ b,
                          bf16_t* __restrict__ xn) {
    const int gwave = (blockIdx.x * blockDim.x + threadIdx.x) >> 5;
    const int lane  = threadIdx.x & 31;
    const float* xp = x + (size_t)gwave * CZ + lane * 4;
    float4 xv = *reinterpret_cast<const float4*>(xp);

    float s = xv.x + xv.y + xv.z + xv.w;
#pragma unroll
    for (int m = 16; m >= 1; m >>= 1) s += __shfl_xor(s, m, 32);
    const float mu = s * (1.0f / 128.0f);

    float d0 = xv.x - mu, d1 = xv.y - mu, d2 = xv.z - mu, d3 = xv.w - mu;
    float sq = d0 * d0 + d1 * d1 + d2 * d2 + d3 * d3;
#pragma unroll
    for (int m = 16; m >= 1; m >>= 1) sq += __shfl_xor(sq, m, 32);
    const float rs = rsqrtf(sq * (1.0f / 128.0f) + EPSLN);

    float4 wv = *reinterpret_cast<const float4*>(w + lane * 4);
    float4 bv = *reinterpret_cast<const float4*>(b + lane * 4);
    v4bf o;
    o[0] = (bf16_t)(d0 * rs * wv.x + bv.x);
    o[1] = (bf16_t)(d1 * rs * wv.y + bv.y);
    o[2] = (bf16_t)(d2 * rs * wv.z + bv.z);
    o[3] = (bf16_t)(d3 * rs * wv.w + bv.w);
    *reinterpret_cast<v4bf*>(xn + (size_t)gwave * CZ + lane * 4) = o;
}

// ---------------------------------------------------------------------------
// Kernel 2: tb[h][p] = sum_c xn[p][c] * wz[c][h]
// ---------------------------------------------------------------------------
__global__ void tb_kernel(const bf16_t* __restrict__ xn,
                          const float* __restrict__ wz,
                          float* __restrict__ tb) {
    const int p = blockIdx.x * 256 + threadIdx.x;
    const uint4* xq = reinterpret_cast<const uint4*>(xn + (size_t)p * CZ);
    float a0 = 0.f, a1 = 0.f, a2 = 0.f, a3 = 0.f;
#pragma unroll
    for (int c8 = 0; c8 < 16; ++c8) {
        Pack8 u; u.q = xq[c8];
#pragma unroll
        for (int i = 0; i < 8; ++i) {
            const float xv = (float)u.e[i];
            const int c = c8 * 8 + i;
            a0 += xv * wz[c * NHEADS + 0];
            a1 += xv * wz[c * NHEADS + 1];
            a2 += xv * wz[c * NHEADS + 2];
            a3 += xv * wz[c * NHEADS + 3];
        }
    }
    tb[0 * (size_t)PTOT + p] = a0;
    tb[1 * (size_t)PTOT + p] = a1;
    tb[2 * (size_t)PTOT + p] = a2;
    tb[3 * (size_t)PTOT + p] = a3;
}

// ---------------------------------------------------------------------------
// Kernel 3/5: WMMA GEMM  out[M=PTOT x 128] = X[M x 128] * W[128 x 128]
// Block = 256 threads (8 waves), tile 128(M) x 128(N), K=128.
// X tile staged to LDS by the Tensor Data Mover (async DMA, TENSORcnt).
// mode: 0 = bf16 out, 1 = bf16 out in V-transposed layout, 2 = f32 out
// ---------------------------------------------------------------------------
__global__ void gemm128_kernel(const bf16_t* __restrict__ X,
                               const bf16_t* __restrict__ WT,
                               const float* __restrict__ pbias, // nullable
                               void* __restrict__ outp,
                               int mode, int do_sigmoid, float scale) {
    __shared__ bf16_t sX[128 * 128];   // 32 KB

    const int tid   = threadIdx.x;
    const size_t mb = (size_t)blockIdx.x * 128;

#ifdef USE_TDM
    // Tensor DMA: 128x128 x 2B tile, global -> LDS, one wave issues it.
    if (tid < 32) {
        const unsigned long long ga = (unsigned long long)(uintptr_t)(X + mb * CZ);
        u32x4 g0;
        g0[0] = 1u;                                   // count=1, user descriptor
        g0[1] = (unsigned int)(uintptr_t)(&sX[0]);    // lds_addr (bytes)
        g0[2] = (unsigned int)ga;                     // global_addr[31:0]
        g0[3] = ((unsigned int)(ga >> 32) & 0x01FFFFFFu) | (2u << 30); // [56:32] + type=2
        i32x8 g1;
        g1[0] = 0x00010000;        // workgroup_mask=0, data_size=1 (2 bytes)
        g1[1] = (int)(128u << 16); // tensor_dim0 = 128
        g1[2] = (int)(128u << 16); // tensor_dim1 = 128
        g1[3] = (int)(128u << 16); // tile_dim0 = 128
        g1[4] = 128;               // tile_dim1 = 128, tile_dim2 = 0
        g1[5] = 128;               // tensor_dim0_stride = 128
        g1[6] = 0;
        g1[7] = 0;
        i32x4 z4; z4[0] = 0; z4[1] = 0; z4[2] = 0; z4[3] = 0;
#if __clang_major__ >= 23
        i32x8 z8; for (int i = 0; i < 8; ++i) z8[i] = 0;
        __builtin_amdgcn_tensor_load_to_lds(g0, g1, z4, z4, z8, 0);
#else
        __builtin_amdgcn_tensor_load_to_lds(g0, g1, z4, z4, 0);
#endif
        __builtin_amdgcn_s_wait_tensorcnt(0);
    }
    __syncthreads();
#else
    // fallback: coalesced 16B copies
    {
        const uint4* gx = reinterpret_cast<const uint4*>(X + mb * CZ);
        uint4* lx = reinterpret_cast<uint4*>(sX);
#pragma unroll
        for (int i = 0; i < 8; ++i) lx[tid + i * 256] = gx[tid + i * 256];
    }
    __syncthreads();
#endif

    const int lane = tid & 31, wave = tid >> 5;
    const int half = lane >> 4, col = lane & 15;
    const int m0 = wave * 16;

    // 4 A fragments (16x32) via ds_load_b128 pairs
    FragBF afr[4];
    const bf16_t* ax = &sX[(m0 + col) * 128];
#pragma unroll
    for (int kk = 0; kk < 4; ++kk) {
        const int base = kk * 32 + half * 16;
        afr[kk].q[0] = *reinterpret_cast<const uint4*>(ax + base);
        afr[kk].q[1] = *reinterpret_cast<const uint4*>(ax + base + 8);
    }

    v8f acc[8];
#pragma unroll
    for (int n = 0; n < 8; ++n)
#pragma unroll
        for (int j = 0; j < 8; ++j) acc[n][j] = 0.f;

#pragma unroll
    for (int n = 0; n < 8; ++n) {
#pragma unroll
        for (int kk = 0; kk < 4; ++kk) {
            FragBF b;
            const bf16_t* bw = WT + (size_t)(n * 16 + col) * 128 + kk * 32 + half * 16;
            b.q[0] = *reinterpret_cast<const uint4*>(bw);
            b.q[1] = *reinterpret_cast<const uint4*>(bw + 8);
            acc[n] = wmma_bf16(afr[kk].v, b.v, acc[n]);
        }
    }

    // epilogue
    const int r_blk   = (int)(mb / NPOS);        // tile never crosses a row r
    const int key_blk = (int)(mb % NPOS);
#pragma unroll
    for (int n = 0; n < 8; ++n) {
        const int ncol = n * 16 + col;
        const float bias = pbias ? pbias[ncol] : 0.f;
#pragma unroll
        for (int j = 0; j < 8; ++j) {
            float v = acc[n][j] * scale + bias;
            if (do_sigmoid) v = 1.0f / (1.0f + __expf(-v));
            const int mloc = m0 + half * 8 + j;
            if (mode == 2) {
                reinterpret_cast<float*>(outp)[(mb + mloc) * 128 + ncol] = v;
            } else if (mode == 1) {
                // vt[ncol][r][key], key contiguous
                const size_t idx = ((size_t)ncol * NPOS + r_blk) * NPOS + key_blk + mloc;
                reinterpret_cast<bf16_t*>(outp)[idx] = (bf16_t)v;
            } else {
                reinterpret_cast<bf16_t*>(outp)[(mb + mloc) * 128 + ncol] = (bf16_t)v;
            }
        }
    }
}

// ---------------------------------------------------------------------------
// Kernel 4: flash attention per (r, h).  Wave handles 16 q-rows.
// grid = (3, NHEADS, NPOS), block = 256 (8 waves -> 128 q rows / block).
// ---------------------------------------------------------------------------
__global__ void attn_kernel(const bf16_t* __restrict__ qbuf,
                            const bf16_t* __restrict__ kbuf,
                            const bf16_t* __restrict__ vt,   // [128][384][384]
                            const bf16_t* __restrict__ gbuf,
                            const float*  __restrict__ tb,
                            bf16_t* __restrict__ og) {
    __shared__ bf16_t sP[8 * 16 * 32];   // per-wave 16x32 P staging (8 KB)

    const int tid  = threadIdx.x;
    const int lane = tid & 31, wave = tid >> 5;
    const int half = lane >> 4, col = lane & 15;
    const int q0   = blockIdx.x * 128 + wave * 16;
    const int h    = blockIdx.y;
    const int r    = blockIdx.z;
    const size_t rowbase = (size_t)r * NPOS;
    const int b0 = half * 16;

    // Q fragment (16x32, already scaled)
    FragBF qf;
    const bf16_t* qp = qbuf + (rowbase + q0 + col) * CZ + h * CHID;
    qf.q[0] = *reinterpret_cast<const uint4*>(qp + b0);
    qf.q[1] = *reinterpret_cast<const uint4*>(qp + b0 + 8);

    v8f o1, o2;
    float m[8], l[8];
#pragma unroll
    for (int j = 0; j < 8; ++j) { o1[j] = 0.f; o2[j] = 0.f; m[j] = -3.0e38f; l[j] = 0.f; }

    const float* tbh = tb + (size_t)h * PTOT;
    bf16_t* pst = sP + wave * (16 * 32);
    // V^T row pointers for this (r,h): element (d, key)
    const bf16_t* vtd1 = vt + ((size_t)(h * CHID + col)      * NPOS + r) * NPOS;
    const bf16_t* vtd2 = vt + ((size_t)(h * CHID + col + 16) * NPOS + r) * NPOS;

    for (int kt = 0; kt < 12; ++kt) {
        const int k0 = kt * 32;

        // K^T fragments: B(d, key) = K[key][d], d contiguous
        FragBF bk1, bk2;
        const bf16_t* kp1 = kbuf + (rowbase + k0 + col) * CZ + h * CHID + half * 16;
        const bf16_t* kp2 = kp1 + 16 * CZ;
        bk1.q[0] = *reinterpret_cast<const uint4*>(kp1);
        bk1.q[1] = *reinterpret_cast<const uint4*>(kp1 + 8);
        bk2.q[0] = *reinterpret_cast<const uint4*>(kp2);
        bk2.q[1] = *reinterpret_cast<const uint4*>(kp2 + 8);

        v8f z;
#pragma unroll
        for (int j = 0; j < 8; ++j) z[j] = 0.f;
        v8f s1 = wmma_bf16(qf.v, bk1.v, z);
        v8f s2 = wmma_bf16(qf.v, bk2.v, z);

        // + pair bias tb[h][q][key]
#pragma unroll
        for (int j = 0; j < 8; ++j) {
            const size_t trow = (size_t)(q0 + half * 8 + j) * NPOS + k0 + col;
            s1[j] += tbh[trow];
            s2[j] += tbh[trow + 16];
        }

        // online softmax
        float mt[8];
#pragma unroll
        for (int j = 0; j < 8; ++j) mt[j] = fmaxf(s1[j], s2[j]);
#pragma unroll
        for (int msk = 8; msk >= 1; msk >>= 1)
#pragma unroll
            for (int j = 0; j < 8; ++j) mt[j] = fmaxf(mt[j], __shfl_xor(mt[j], msk, 32));

        float rs[8];
#pragma unroll
        for (int j = 0; j < 8; ++j) {
            const float mn = fmaxf(m[j], mt[j]);
            const float alpha = __expf(m[j] - mn);
            m[j] = mn;
            o1[j] *= alpha; o2[j] *= alpha; l[j] *= alpha;
            s1[j] = __expf(s1[j] - mn);
            s2[j] = __expf(s2[j] - mn);
            rs[j] = s1[j] + s2[j];
        }
#pragma unroll
        for (int msk = 8; msk >= 1; msk >>= 1)
#pragma unroll
            for (int j = 0; j < 8; ++j) rs[j] += __shfl_xor(rs[j], msk, 32);
#pragma unroll
        for (int j = 0; j < 8; ++j) l[j] += rs[j];

        // transpose P (C-frag) -> A-frag via wave-private LDS
#pragma unroll
        for (int j = 0; j < 8; ++j) {
            const int rr = half * 8 + j;
            pst[rr * 32 + col]      = (bf16_t)s1[j];
            pst[rr * 32 + 16 + col] = (bf16_t)s2[j];
        }
        FragBF pf;
        const bf16_t* pr = pst + col * 32;
        pf.q[0] = *reinterpret_cast<const uint4*>(pr + b0);
        pf.q[1] = *reinterpret_cast<const uint4*>(pr + b0 + 8);

        // V fragments from V^T: B(key, d), key contiguous
        FragBF bv1, bv2;
        bv1.q[0] = *reinterpret_cast<const uint4*>(vtd1 + k0 + half * 16);
        bv1.q[1] = *reinterpret_cast<const uint4*>(vtd1 + k0 + half * 16 + 8);
        bv2.q[0] = *reinterpret_cast<const uint4*>(vtd2 + k0 + half * 16);
        bv2.q[1] = *reinterpret_cast<const uint4*>(vtd2 + k0 + half * 16 + 8);

        o1 = wmma_bf16(pf.v, bv1.v, o1);
        o2 = wmma_bf16(pf.v, bv2.v, o2);
    }

    // normalize, gate, store
#pragma unroll
    for (int j = 0; j < 8; ++j) {
        const float inv = 1.0f / l[j];
        const int qrow = q0 + half * 8 + j;
        const size_t base = (rowbase + qrow) * CZ + h * CHID;
        const float g1 = (float)gbuf[base + col];
        const float g2 = (float)gbuf[base + 16 + col];
        og[base + col]      = (bf16_t)(o1[j] * inv * g1);
        og[base + 16 + col] = (bf16_t)(o2[j] * inv * g2);
    }
}

// ---------------------------------------------------------------------------
// Launch
// ---------------------------------------------------------------------------
extern "C" void kernel_launch(void* const* d_in, const int* in_sizes, int n_in,
                              void* d_out, int out_size, void* d_ws, size_t ws_size,
                              hipStream_t stream) {
    const float* x    = (const float*)d_in[0];
    const float* ln_w = (const float*)d_in[1];
    const float* ln_b = (const float*)d_in[2];
    const float* wz   = (const float*)d_in[3];
    const float* wq   = (const float*)d_in[4];
    const float* wk   = (const float*)d_in[5];
    const float* wv   = (const float*)d_in[6];
    const float* wg   = (const float*)d_in[7];
    const float* bg   = (const float*)d_in[8];
    const float* wo   = (const float*)d_in[9];
    const float* bo   = (const float*)d_in[10];
    float* out = (float*)d_out;

    char* ws = (char*)d_ws;
    const size_t SZ_BF = (size_t)PTOT * CZ * sizeof(bf16_t);   // 36 MiB each
    bf16_t* xn  = (bf16_t*)(ws);
    bf16_t* qb  = (bf16_t*)(ws + 1 * SZ_BF);
    bf16_t* kb  = (bf16_t*)(ws + 2 * SZ_BF);
    bf16_t* vtb = (bf16_t*)(ws + 3 * SZ_BF);   // V transposed [128][384][384]
    bf16_t* gb  = (bf16_t*)(ws + 4 * SZ_BF);
    bf16_t* ob  = (bf16_t*)(ws + 5 * SZ_BF);
    float*  tb  = (float*)(ws + 6 * SZ_BF);    // 4*PTOT f32
    bf16_t* wts = (bf16_t*)(ws + 6 * SZ_BF + (size_t)4 * PTOT * sizeof(float));
    bf16_t* wtq = wts + 0 * 16384;
    bf16_t* wtk = wts + 1 * 16384;
    bf16_t* wtv = wts + 2 * 16384;
    bf16_t* wtg = wts + 3 * 16384;
    bf16_t* wto = wts + 4 * 16384;

    // 0) weight prep: f32 row-major -> bf16 col-major
    wt_kernel<<<64, 256, 0, stream>>>(wq, wtq);
    wt_kernel<<<64, 256, 0, stream>>>(wk, wtk);
    wt_kernel<<<64, 256, 0, stream>>>(wv, wtv);
    wt_kernel<<<64, 256, 0, stream>>>(wg, wtg);
    wt_kernel<<<64, 256, 0, stream>>>(wo, wto);

    // 1) LayerNorm -> bf16
    ln_kernel<<<PTOT / 8, 256, 0, stream>>>(x, ln_w, ln_b, xn);

    // 2) pair bias tb = xn * wz
    tb_kernel<<<PTOT / 256, 256, 0, stream>>>(xn, wz, tb);

    // 3) projections (WMMA GEMMs, TDM-staged X tiles)
    const int GB = PTOT / 128;   // 1152 blocks
    gemm128_kernel<<<GB, 256, 0, stream>>>(xn, wtq, nullptr, qb,  0, 0, QSCALE);
    gemm128_kernel<<<GB, 256, 0, stream>>>(xn, wtk, nullptr, kb,  0, 0, 1.0f);
    gemm128_kernel<<<GB, 256, 0, stream>>>(xn, wtv, nullptr, vtb, 1, 0, 1.0f);
    gemm128_kernel<<<GB, 256, 0, stream>>>(xn, wtg, bg,      gb,  0, 1, 1.0f);

    // 4) flash attention + gate (WMMA)
    attn_kernel<<<dim3(3, NHEADS, NPOS), 256, 0, stream>>>(qb, kb, vtb, gb, tb, ob);

    // 5) output projection (WMMA, f32 out + bo)
    gemm128_kernel<<<GB, 256, 0, stream>>>(ob, wto, bo, out, 2, 0, 1.0f);
}
